// DeformableHistoryAttention_4148938408691
// MI455X (gfx1250) — matrix operand order repaired and checked by reading, not compiled, not run on hardware
//
#include <hip/hip_runtime.h>
#include <hip/hip_bf16.h>
#include <math.h>

// ---------------- types ----------------
typedef __attribute__((ext_vector_type(16))) _Float16 v16h;
typedef __attribute__((ext_vector_type(8)))  _Float16 v8h;
typedef __attribute__((ext_vector_type(4)))  _Float16 v4h;
typedef __attribute__((ext_vector_type(8)))  float    v8f;

// ---------------- problem constants ----------------
#define BSZ   4
#define SEQ   4096
#define EDIM  512
#define HEADS 8
#define NPTS  16
#define MROWS (BSZ * SEQ)        // 16384
#define KDIM  EDIM               // 512 (K of every GEMM)

// GEMM tiling
#define BM    128
#define BN    32
#define KPAD  520                // column pad (halves) -> conflict-free ds reads

// ---------------- f32 -> f16 convert (vectorized x4) ----------------
__global__ __launch_bounds__(256) void cvt_f32_f16_x4(const float* __restrict__ in,
                                                      _Float16* __restrict__ out, int n4) {
  int i = blockIdx.x * 256 + threadIdx.x;
  if (i < n4) {
    float4 f = ((const float4*)in)[i];
    v4h h;
    h[0] = (_Float16)f.x; h[1] = (_Float16)f.y;
    h[2] = (_Float16)f.z; h[3] = (_Float16)f.w;
    ((v4h*)out)[i] = h;
  }
}

// ---------------- valid_len reduction over padding mask ----------------
__global__ __launch_bounds__(256) void validlen_kernel(const unsigned char* __restrict__ mask,
                                                       int* __restrict__ vlen) {
  __shared__ int red[256];
  int b = blockIdx.x;
  int tsum = 0;
  for (int i = threadIdx.x; i < SEQ; i += 256)
    tsum += (mask[(size_t)b * SEQ + i] != 0) ? 1 : 0;
  red[threadIdx.x] = tsum;
  __syncthreads();
  for (int st = 128; st > 0; st >>= 1) {
    if (threadIdx.x < st) red[threadIdx.x] += red[threadIdx.x + st];
    __syncthreads();
  }
  if (threadIdx.x == 0) {
    int v = SEQ - red[0];
    vlen[b] = v < 1 ? 1 : v;
  }
}

// ---------------- WMMA GEMM:  C[M,N] = act(A[M,512]_f16 @ W[512,N]_f16 + bias) ----------------
// ACT: 0=none 1=gelu(exact) 2=tanh   OUTH: 1 -> f16 out, 0 -> f32 out
template <int ACT, int OUTH>
__global__ __launch_bounds__(256) void gemm_wmma_f16(const _Float16* __restrict__ A,
                                                     const _Float16* __restrict__ W,
                                                     const float* __restrict__ bias,
                                                     float* __restrict__ outF,
                                                     _Float16* __restrict__ outH,
                                                     int N) {
  __shared__ _Float16 ldsW[BN * KPAD];   // 33,280 B: W tile, column-major, padded

  const int tid   = threadIdx.x;
  const int wave  = tid >> 5;
  const int lane  = tid & 31;
  const int g     = lane >> 4;           // lane group (0: lanes 0-15, 1: lanes 16-31)
  const int ln    = lane & 15;
  const int mbase = blockIdx.x * BM + wave * 16;
  const int nbase = blockIdx.y * BN;

  // Stage W[:, nbase:nbase+32] transposed into LDS (coalesced global reads).
  for (int i = tid; i < KDIM * BN; i += 256) {
    int k = i >> 5;        // / BN
    int n = i & (BN - 1);
    ldsW[n * KPAD + k] = W[(size_t)k * N + nbase + n];
  }
  __syncthreads();

  v8f zero = {0.f, 0.f, 0.f, 0.f, 0.f, 0.f, 0.f, 0.f};
  v8f acc0 = zero, acc1 = zero;

  // A fragment base: lanes 0-15 -> row m, halves start K; lanes 16-31 -> +8 (ISA 16-bit A layout)
  const _Float16* arow = A + (size_t)(mbase + ln) * KDIM + 8 * g;
  __builtin_prefetch(arow, 0, 0);   // global_prefetch_b8

#pragma unroll
  for (int ks = 0; ks < KDIM; ks += 32) {
    v8h alo = *(const v8h*)(arow + ks);        // K: ks+8g   .. +7
    v8h ahi = *(const v8h*)(arow + ks + 16);   // K: ks+16+8g.. +7
    v16h a;
#pragma unroll
    for (int i = 0; i < 8; ++i) { a[i] = alo[i]; a[i + 8] = ahi[i]; }

    // B fragment, tile 0 (cols nbase..+15): lane (g,ln) holds K = ks+8g+0..7, ks+16+8g+0..7 of col ln
    const _Float16* bp0 = &ldsW[(0 * 16 + ln) * KPAD + ks + 8 * g];
    v8h b0l = *(const v8h*)bp0;
    v8h b0h = *(const v8h*)(bp0 + 16);
    v16h b0;
#pragma unroll
    for (int i = 0; i < 8; ++i) { b0[i] = b0l[i]; b0[i + 8] = b0h[i]; }
    acc0 = __builtin_amdgcn_wmma_f32_16x16x32_f16(false, a, false, b0, (short)0, acc0,
                                                  false, false);

    // B fragment, tile 1 (cols nbase+16..+31)
    const _Float16* bp1 = &ldsW[(1 * 16 + ln) * KPAD + ks + 8 * g];
    v8h b1l = *(const v8h*)bp1;
    v8h b1h = *(const v8h*)(bp1 + 16);
    v16h b1;
#pragma unroll
    for (int i = 0; i < 8; ++i) { b1[i] = b1l[i]; b1[i + 8] = b1h[i]; }
    acc1 = __builtin_amdgcn_wmma_f32_16x16x32_f16(false, a, false, b1, (short)0, acc1,
                                                  false, false);
  }

  // Epilogue. C/D layout: VGPR r holds M = r + 8*g, N = lane&15.
#pragma unroll
  for (int nt = 0; nt < 2; ++nt) {
    int n  = nbase + nt * 16 + ln;
    float bv = bias[n];
    const v8f accv = nt == 0 ? acc0 : acc1;
#pragma unroll
    for (int r = 0; r < 8; ++r) {
      int m = mbase + 8 * g + r;
      float val = accv[r] + bv;
      if (ACT == 1) val = 0.5f * val * (1.0f + erff(val * 0.70710678118654752f));
      else if (ACT == 2) val = tanhf(val);
      if (OUTH) outH[(size_t)m * N + n] = (_Float16)val;
      else      outF[(size_t)m * N + n] = val;
    }
  }
}

// ---------------- deformable gather-attention: one wave per query ----------------
__global__ __launch_bounds__(256) void attn_kernel(const float* __restrict__ T,      // [M,128] tanh outputs
                                                   const _Float16* __restrict__ Q,   // [M,512]
                                                   const _Float16* __restrict__ Kk,  // [M,512]
                                                   const _Float16* __restrict__ V,   // [M,512]
                                                   const unsigned char* __restrict__ mask,
                                                   const int* __restrict__ vlen,
                                                   const float* __restrict__ anchors,
                                                   _Float16* __restrict__ Out)       // [M,512]
{
  const int lane = threadIdx.x & 31;
  const int wave = threadIdx.x >> 5;
  const int gq   = blockIdx.x * 8 + wave;
  const int b    = gq >> 12;    // / 4096
  const int s    = gq & (SEQ - 1);

  // --- sample positions: lane p (0..15, duplicated in upper half) ---
  const int p = lane & 15;
  const float* trow = T + (size_t)gq * (HEADS * NPTS);
  float osum = 0.f;                           // mean over H * OFFSET_SCALE(8) == plain sum
#pragma unroll
  for (int h = 0; h < HEADS; ++h) osum += trow[h * NPTS + p];
  float pos  = (float)s;
  float samp = anchors[p] * pos + osum;
  samp = fminf(fmaxf(samp, fmaxf(pos - 1024.0f, 0.0f)), pos);
  samp = fminf(samp, (float)(vlen[b] - 1));
  int idx = (int)__builtin_rintf(samp);       // round-half-even == jnp.round
  int inv = (mask[(size_t)b * SEQ + idx] != 0) || (idx > s);

  // --- scores: 2 lanes per point, 256 elements each ---
  const int p2   = lane >> 1;
  const int half = lane & 1;
  int idxp = __shfl(idx, p2);
  int invp = __shfl(inv, p2);
  const _Float16* qrow = Q + (size_t)gq * EDIM + half * 256;
  const _Float16* krow = Kk + ((size_t)b * SEQ + idxp) * EDIM + half * 256;
  float acc = 0.f;
#pragma unroll
  for (int i = 0; i < 32; ++i) {
    v8h qv = *(const v8h*)(qrow + i * 8);
    v8h kv = *(const v8h*)(krow + i * 8);
#pragma unroll
    for (int j = 0; j < 8; ++j) acc += (float)qv[j] * (float)kv[j];
  }
  acc += __shfl_xor(acc, 1);                  // lanes 2p,2p+1 both hold full dot
  float score = acc * 0.04419417382415922f;   // 1/sqrt(512)

  int all_inv = __all(invp);
  if (invp) score = -3.0e38f;

  // softmax over the 16 points (values duplicated per lane pair)
  float mx = score;
  mx = fmaxf(mx, __shfl_xor(mx, 2));
  mx = fmaxf(mx, __shfl_xor(mx, 4));
  mx = fmaxf(mx, __shfl_xor(mx, 8));
  mx = fmaxf(mx, __shfl_xor(mx, 16));
  float e = invp ? 0.f : __expf(score - mx);
  float se = e;
  se += __shfl_xor(se, 2);
  se += __shfl_xor(se, 4);
  se += __shfl_xor(se, 8);
  se += __shfl_xor(se, 16);                   // == 2 * sum
  float attnp = (all_inv || se <= 0.f) ? 0.f : (e * 2.0f) / se;

  // --- weighted sum of gathered V rows: lane owns channels [16*lane, 16*lane+16) ---
  float oacc[16];
#pragma unroll
  for (int i = 0; i < 16; ++i) oacc[i] = 0.f;
  const _Float16* vbase = V + (size_t)b * SEQ * EDIM;
#pragma unroll
  for (int pp = 0; pp < 16; ++pp) {
    float aw = __shfl(attnp, 2 * pp);         // wave-uniform
    int   ip = __shfl(idxp, 2 * pp);
    if (aw != 0.0f) {
      const _Float16* vrow = vbase + (size_t)ip * EDIM + lane * 16;
      v8h v0 = *(const v8h*)(vrow);
      v8h v1 = *(const v8h*)(vrow + 8);
#pragma unroll
      for (int j = 0; j < 8; ++j) {
        oacc[j]     += aw * (float)v0[j];
        oacc[j + 8] += aw * (float)v1[j];
      }
    }
  }
  _Float16* orow = Out + (size_t)gq * EDIM + lane * 16;
  v8h o0, o1;
#pragma unroll
  for (int j = 0; j < 8; ++j) { o0[j] = (_Float16)oacc[j]; o1[j] = (_Float16)oacc[j + 8]; }
  *(v8h*)orow       = o0;
  *(v8h*)(orow + 8) = o1;
}

// ---------------- host-side orchestration ----------------
extern "C" void kernel_launch(void* const* d_in, const int* in_sizes, int n_in,
                              void* d_out, int out_size, void* d_ws, size_t ws_size,
                              hipStream_t stream) {
  (void)in_sizes; (void)n_in; (void)out_size; (void)ws_size;

  const float*         x       = (const float*)d_in[0];
  const unsigned char* mask    = (const unsigned char*)d_in[1];
  const float* Wq = (const float*)d_in[2];  const float* bq = (const float*)d_in[3];
  const float* Wk = (const float*)d_in[4];  const float* bk = (const float*)d_in[5];
  const float* Wv = (const float*)d_in[6];  const float* bv = (const float*)d_in[7];
  const float* Wo = (const float*)d_in[8];  const float* bo = (const float*)d_in[9];
  const float* W1 = (const float*)d_in[10]; const float* b1 = (const float*)d_in[11];
  const float* W2 = (const float*)d_in[12]; const float* b2 = (const float*)d_in[13];
  const float* anchors = (const float*)d_in[14];

  // workspace carve-up
  char* ws = (char*)d_ws;
  size_t off = 0;
  auto carve = [&](size_t bytes) -> void* {
    void* pp = ws + off;
    off = (off + bytes + 255) & ~(size_t)255;
    return pp;
  };
  const size_t actH = (size_t)MROWS * EDIM * sizeof(_Float16);   // 16 MB
  _Float16* xh  = (_Float16*)carve(actH);
  _Float16* w1h = (_Float16*)carve((size_t)EDIM * EDIM * 2);
  _Float16* wqh = (_Float16*)carve((size_t)EDIM * EDIM * 2);
  _Float16* wkh = (_Float16*)carve((size_t)EDIM * EDIM * 2);
  _Float16* wvh = (_Float16*)carve((size_t)EDIM * EDIM * 2);
  _Float16* woh = (_Float16*)carve((size_t)EDIM * EDIM * 2);
  _Float16* w2h = (_Float16*)carve((size_t)EDIM * HEADS * NPTS * 2);
  _Float16* hh  = (_Float16*)carve(actH);                        // gelu(xW1+b1); reused as attn-out
  float*    tt  = (float*)   carve((size_t)MROWS * HEADS * NPTS * 4);
  _Float16* qh  = (_Float16*)carve(actH);
  _Float16* kh  = (_Float16*)carve(actH);
  _Float16* vh  = (_Float16*)carve(actH);
  int*      vl  = (int*)     carve(256);
  _Float16* aoh = hh;   // h no longer needed once tt is computed

  // 1) converts
  {
    int n4 = (MROWS * EDIM) / 4;                                 // x
    cvt_f32_f16_x4<<<dim3((n4 + 255) / 256), dim3(256), 0, stream>>>(x, xh, n4);
    int w4 = (EDIM * EDIM) / 4;
    cvt_f32_f16_x4<<<dim3((w4 + 255) / 256), dim3(256), 0, stream>>>(W1, w1h, w4);
    cvt_f32_f16_x4<<<dim3((w4 + 255) / 256), dim3(256), 0, stream>>>(Wq, wqh, w4);
    cvt_f32_f16_x4<<<dim3((w4 + 255) / 256), dim3(256), 0, stream>>>(Wk, wkh, w4);
    cvt_f32_f16_x4<<<dim3((w4 + 255) / 256), dim3(256), 0, stream>>>(Wv, wvh, w4);
    cvt_f32_f16_x4<<<dim3((w4 + 255) / 256), dim3(256), 0, stream>>>(Wo, woh, w4);
    int w24 = (EDIM * HEADS * NPTS) / 4;
    cvt_f32_f16_x4<<<dim3((w24 + 255) / 256), dim3(256), 0, stream>>>(W2, w2h, w24);
  }

  // 2) valid_len
  validlen_kernel<<<dim3(BSZ), dim3(256), 0, stream>>>(mask, vl);

  // 3) GEMMs (grid: M/128 x N/32)
  dim3 blk(256);
  dim3 gE(MROWS / BM, EDIM / BN);                 // (128, 16)
  dim3 gP(MROWS / BM, (HEADS * NPTS) / BN);       // (128, 4)

  gemm_wmma_f16<1, 1><<<gE, blk, 0, stream>>>(xh, w1h, b1, nullptr, hh, EDIM);          // h = gelu
  gemm_wmma_f16<2, 0><<<gP, blk, 0, stream>>>(hh, w2h, b2, tt, nullptr, HEADS * NPTS);  // t = tanh
  gemm_wmma_f16<0, 1><<<gE, blk, 0, stream>>>(xh, wqh, bq, nullptr, qh, EDIM);          // q
  gemm_wmma_f16<0, 1><<<gE, blk, 0, stream>>>(xh, wkh, bk, nullptr, kh, EDIM);          // k
  gemm_wmma_f16<0, 1><<<gE, blk, 0, stream>>>(xh, wvh, bv, nullptr, vh, EDIM);          // v

  // 4) deformable gather attention (one wave per query)
  attn_kernel<<<dim3(MROWS / 8), blk, 0, stream>>>(tt, qh, kh, vh, mask, vl, anchors, aoh);

  // 5) final projection -> fp32 output
  gemm_wmma_f16<0, 0><<<gE, blk, 0, stream>>>(aoh, woh, bo, (float*)d_out, nullptr, EDIM);
}
